// GroupedQueryAttention_18408229830862
// MI455X (gfx1250) — matrix operand order repaired and checked by reading, not compiled, not run on hardware
//
#include <hip/hip_runtime.h>
#include <hip/hip_bf16.h>

#define DIM   2048
#define NSEQ  2048
#define QH    32
#define KVH   8
#define HD    64
#define KVD   512
#define SCALE 0.125f   // 64^-0.5

typedef __bf16 bf16_t;
typedef __attribute__((ext_vector_type(16))) __bf16 v16bf;
typedef __attribute__((ext_vector_type(8)))  __bf16 v8bf;
typedef __attribute__((ext_vector_type(8)))  float  v8f;

#if defined(__gfx1250__) && __has_builtin(__builtin_amdgcn_tensor_load_to_lds)
#define HAVE_TDM 1
typedef __attribute__((ext_vector_type(4))) unsigned int v4ui;
typedef __attribute__((ext_vector_type(8))) int          v8si;
typedef __attribute__((ext_vector_type(4))) int          v4si;
#else
#define HAVE_TDM 0
#endif

// ---- helpers -------------------------------------------------------------
static __device__ __forceinline__ void store_val(float*  p, float v) { *p = v; }
static __device__ __forceinline__ void store_val(bf16_t* p, float v) { *p = (bf16_t)v; }

// load 8 consecutive elements as bf16x8 (b128 global loads)
static __device__ __forceinline__ v8bf load8(const float* p) {
  float4 a = ((const float4*)p)[0];
  float4 b = ((const float4*)p)[1];
  v8bf r;
  r[0] = (bf16_t)a.x; r[1] = (bf16_t)a.y; r[2] = (bf16_t)a.z; r[3] = (bf16_t)a.w;
  r[4] = (bf16_t)b.x; r[5] = (bf16_t)b.y; r[6] = (bf16_t)b.z; r[7] = (bf16_t)b.w;
  return r;
}
static __device__ __forceinline__ v8bf load8(const bf16_t* p) { return *(const v8bf*)p; }

// A-fragment (16x32 bf16): lane row = ln, K = {16hi..16hi+15}
// => one contiguous 32B run in a row-major tile row (chunk base = row + ch*32)
static __device__ __forceinline__ v16bf frag_a(const bf16_t* chunkRow, int hi) {
  return *(const v16bf*)(chunkRow + 16 * hi);
}

static __device__ __forceinline__ v8f wmma_bf16(const v16bf& a, const v16bf& b, const v8f& c) {
  return __builtin_amdgcn_wmma_f32_16x16x32_bf16(false, a, false, b, (short)0, c, false, false);
}

// ---- generic GEMM:  C[M,N] = A[M,K] * W[K,N] + bias ----------------------
// block tile 64(M) x 128(N), 256 threads = 8 waves, K-step 64 (8 WMMA/stage)
template <typename TA, typename TO>
__global__ __launch_bounds__(256)
void gemm_bias_kernel(const TA* __restrict__ A, const float* __restrict__ W,
                      const float* __restrict__ bias, TO* __restrict__ C,
                      int M, int N, int K) {
  __shared__ __align__(32) bf16_t As[64][64];     // raw row-major A slice [m][k]
  __shared__ __align__(32) bf16_t Bst[128][64];   // W slice transposed [n][k]

  const int tid  = threadIdx.x;
  const int lane = tid & 31;
  const int wv   = tid >> 5;
  const int tm   = wv & 3;            // tile row within block (0..3)
  const int tnb  = (wv >> 2) * 4;     // first of 4 tile cols (0 or 4)
  const int hi   = lane >> 4;
  const int ln   = lane & 15;
  const int rowBase = blockIdx.y * 64;
  const int colBase = blockIdx.x * 128;

  v8f acc[4];
#pragma unroll
  for (int t = 0; t < 4; ++t)
#pragma unroll
    for (int i = 0; i < 8; ++i) acc[t][i] = 0.0f;

  for (int k0 = 0; k0 < K; k0 += 64) {
    // stage A 64x64 (row-major, b128 reads + b128 LDS writes)
    for (int g = tid; g < 512; g += 256) {
      int r = g >> 3, c8 = (g & 7) << 3;
      *(v8bf*)&As[r][c8] = load8(&A[(size_t)(rowBase + r) * K + (k0 + c8)]);
    }
    // stage W 64x128, transposed into Bst[n][k] (coalesced reads, strided u16 writes)
    for (int g = tid; g < 1024; g += 256) {
      int kr = g >> 4, c8 = (g & 15) << 3;
      v8bf v = load8(&W[(size_t)(k0 + kr) * N + (colBase + c8)]);
#pragma unroll
      for (int i = 0; i < 8; ++i) Bst[c8 + i][kr] = v[i];
    }
    if (k0 + 64 < K) {  // global_prefetch_b8 hints for the next K-slice
      __builtin_prefetch(&A[(size_t)(rowBase + (tid >> 2)) * K + k0 + 64], 0, 2);
      __builtin_prefetch(&W[(size_t)(k0 + 64 + (tid >> 3)) * N + colBase + ((tid & 7) << 4)], 0, 2);
    }
    __syncthreads();

#pragma unroll
    for (int ch = 0; ch < 2; ++ch) {
      v16bf af = frag_a(&As[tm * 16 + ln][ch * 32], hi);
#pragma unroll
      for (int t = 0; t < 4; ++t) {
        // B-fragment: K = e + 16*hi -> one contiguous 32B run in Bst row
        v16bf bfr = *(const v16bf*)&Bst[(tnb + t) * 16 + ln][ch * 32 + 16 * hi];
        acc[t] = wmma_bf16(af, bfr, acc[t]);
      }
    }
    __syncthreads();
  }

#pragma unroll
  for (int t = 0; t < 4; ++t) {
    int col = colBase + (tnb + t) * 16 + ln;
    float bv = bias[col];
#pragma unroll
    for (int r = 0; r < 8; ++r) {
      int row = rowBase + tm * 16 + r + 8 * hi;
      store_val(&C[(size_t)row * N + col], acc[t][r] + bv);
    }
  }
}

// ---- RoPE (interleaved pairs, head_dim 64), in-place on bf16 -------------
__global__ __launch_bounds__(256)
void rope_kernel(bf16_t* __restrict__ X, int ncols) {
  int idx = blockIdx.x * blockDim.x + threadIdx.x;
  int pairsPerRow = ncols >> 1;
  if (idx >= NSEQ * pairsPerRow) return;
  int n  = idx / pairsPerRow;
  int pr = idx - n * pairsPerRow;
  float i2  = (float)((pr & 31) * 2);                    // 2*i within head
  float inv = __powf(10000.0f, -i2 * (1.0f / 64.0f));
  float ang = (float)n * inv;
  float s, c;
  __sincosf(ang, &s, &c);
  size_t base = (size_t)n * ncols + (size_t)pr * 2;
  float x1 = (float)X[base], x2 = (float)X[base + 1];
  X[base]     = (bf16_t)(x1 * c - x2 * s);
  X[base + 1] = (bf16_t)(x1 * s + x2 * c);
}

// ---- flash attention: one WG = (64 query rows, one q-head) ---------------
// 4 waves * 16-row strips; causal; online softmax; bf16 WMMA; TDM K staging
__global__ __launch_bounds__(128)
void attn_kernel(const bf16_t* __restrict__ Q, const bf16_t* __restrict__ Kt,
                 const bf16_t* __restrict__ V, bf16_t* __restrict__ O) {
  __shared__ __align__(32) bf16_t Ks[64][64];      // row-major [key][dim]
  __shared__ __align__(32) bf16_t Vt[64][64];      // transposed [dim][key]
  __shared__ __align__(32) bf16_t Pp[4][16][64];   // per-wave P tile [row][key]

  const int tid  = threadIdx.x;
  const int lane = tid & 31;
  const int wv   = tid >> 5;
  const int hi   = lane >> 4;
  const int ln   = lane & 15;
  const int h    = blockIdx.y;        // q-head
  const int kvh  = h & 7;             // reference reshape: kv-head = h % 8
  const int qblk = blockIdx.x;
  const int qrow0 = qblk * 64 + wv * 16;

  // Q fragments for this wave's 16 rows (two K=32 chunks), pre-scaled
  v16bf qf[2];
  {
    const bf16_t* qrow = &Q[(size_t)(qrow0 + ln) * DIM + h * 64];
#pragma unroll
    for (int ch = 0; ch < 2; ++ch) {
      v8bf lo  = *(const v8bf*)(qrow + ch * 32 + 16 * hi);
      v8bf hi8 = *(const v8bf*)(qrow + ch * 32 + 16 * hi + 8);
#pragma unroll
      for (int i = 0; i < 8; ++i) {
        qf[ch][i]     = (bf16_t)((float)lo[i]  * SCALE);
        qf[ch][8 + i] = (bf16_t)((float)hi8[i] * SCALE);
      }
    }
  }

  v8f o[4];
#pragma unroll
  for (int t = 0; t < 4; ++t)
#pragma unroll
    for (int i = 0; i < 8; ++i) o[t][i] = 0.0f;
  float m[8], l[8];
#pragma unroll
  for (int r = 0; r < 8; ++r) { m[r] = -3.0e38f; l[r] = 0.0f; }

  for (int kb = 0; kb <= qblk; ++kb) {
    __syncthreads();

#if HAVE_TDM
    // Tensor Data Mover: 2D tile load 64(keys) x 64(dims) bf16 -> Ks (row-major)
    if (wv == 0) {
      unsigned lds_off = (unsigned)(size_t)(void*)&Ks[0][0];  // low 32b of flat = LDS offset
      unsigned long long ga =
          (unsigned long long)(size_t)&Kt[(size_t)(kb * 64) * KVD + kvh * 64];
      v4ui g0;
      g0[0] = 1u;                                   // count=1 (valid descriptor)
      g0[1] = lds_off;                              // lds_addr
      g0[2] = (unsigned)(ga & 0xffffffffu);         // global_addr[31:0]
      g0[3] = (unsigned)((ga >> 32) & 0x1ffffffu)   // global_addr[56:32]
              | (2u << 30);                         // type = 2 ("image")
      v8si g1;
      g1[0] = (int)(1u << 16);                      // data_size=1 (2B); mask=0
      g1[1] = (int)((unsigned)KVD << 16);           // tensor_dim0 = 512 (bits 63:48)
      g1[2] = (int)((unsigned)NSEQ << 16);          // tensor_dim1 = 2048 (bits 95:80)
      g1[3] = (int)(64u << 16);                     // tile_dim0 = 64 (bits 127:112)
      g1[4] = 64;                                   // tile_dim1 = 64; tile_dim2 = 0
      g1[5] = KVD;                                  // tensor_dim0_stride = 512
      g1[6] = 0;                                    // stride hi / dim1_stride lo
      g1[7] = 0;
      v4si z4; z4[0] = 0; z4[1] = 0; z4[2] = 0; z4[3] = 0;
      v8si z8;
#pragma unroll
      for (int i = 0; i < 8; ++i) z8[i] = 0;
      __builtin_amdgcn_tensor_load_to_lds(g0, g1, z4, z4, z8, 0);
      __builtin_amdgcn_s_wait_tensorcnt(0);
    }
    // V staged cooperatively (needs transpose), overlapped with the TDM load
    for (int g = tid; g < 512; g += 128) {
      int key = g >> 3, d8 = (g & 7) << 3;
      v8bf vv = *(const v8bf*)&V[(size_t)(kb * 64 + key) * KVD + kvh * 64 + d8];
#pragma unroll
      for (int i = 0; i < 8; ++i) Vt[d8 + i][key] = vv[i];
    }
#else
    for (int g = tid; g < 512; g += 128) {
      int key = g >> 3, d8 = (g & 7) << 3;
      size_t gb = (size_t)(kb * 64 + key) * KVD + kvh * 64 + d8;
      *(v8bf*)&Ks[key][d8] = *(const v8bf*)&Kt[gb];
      v8bf vv = *(const v8bf*)&V[gb];
#pragma unroll
      for (int i = 0; i < 8; ++i) Vt[d8 + i][key] = vv[i];
    }
#endif
    __syncthreads();

    // S = Q * K^T  (16 x 64 per wave); B-frag = contiguous 32B run of Ks row
    float sv[4][8];
#pragma unroll
    for (int tn = 0; tn < 4; ++tn) {
      v8f s;
#pragma unroll
      for (int i = 0; i < 8; ++i) s[i] = 0.0f;
#pragma unroll
      for (int ch = 0; ch < 2; ++ch) {
        v16bf bfr = *(const v16bf*)&Ks[tn * 16 + ln][ch * 32 + 16 * hi];
        s = wmma_bf16(qf[ch], bfr, s);
      }
      int keyg = kb * 64 + tn * 16 + ln;
#pragma unroll
      for (int r = 0; r < 8; ++r) {
        int rowg = qrow0 + r + 8 * hi;
        sv[tn][r] = (keyg <= rowg) ? s[r] : -3.0e38f;
      }
    }

    // online softmax: per-row max/sum via 16-lane xor-shuffle reductions
#pragma unroll
    for (int r = 0; r < 8; ++r) {
      float tmax = fmaxf(fmaxf(sv[0][r], sv[1][r]), fmaxf(sv[2][r], sv[3][r]));
#pragma unroll
      for (int d = 1; d < 16; d <<= 1) tmax = fmaxf(tmax, __shfl_xor(tmax, d, 32));
      float mn = fmaxf(m[r], tmax);
      float alpha = __expf(m[r] - mn);
      m[r] = mn;
      l[r] *= alpha;
#pragma unroll
      for (int t = 0; t < 4; ++t) o[t][r] *= alpha;
      float ps = 0.0f;
#pragma unroll
      for (int t = 0; t < 4; ++t) {
        float p = __expf(sv[t][r] - mn);
        sv[t][r] = p;
        ps += p;
      }
#pragma unroll
      for (int d = 1; d < 16; d <<= 1) ps += __shfl_xor(ps, d, 32);
      l[r] += ps;
    }

    // P: C-layout -> row-major per-wave LDS tile (same-wave DS ops are in-order)
#pragma unroll
    for (int t = 0; t < 4; ++t)
#pragma unroll
      for (int r = 0; r < 8; ++r) Pp[wv][r + 8 * hi][t * 16 + ln] = (bf16_t)sv[t][r];

    // O += P * V ; A-frag = 32B run of Pp row, B-frag = 32B run of Vt row
#pragma unroll
    for (int ch = 0; ch < 2; ++ch) {
      v16bf pa = frag_a(&Pp[wv][ln][ch * 32], hi);
#pragma unroll
      for (int to = 0; to < 4; ++to) {
        v16bf bfr = *(const v16bf*)&Vt[to * 16 + ln][ch * 32 + 16 * hi];
        o[to] = wmma_bf16(pa, bfr, o[to]);
      }
    }
  }

  // normalize and store (diagonal block guarantees l > 0)
#pragma unroll
  for (int t = 0; t < 4; ++t)
#pragma unroll
    for (int r = 0; r < 8; ++r) {
      int rowg = qrow0 + r + 8 * hi;
      float val = o[t][r] / l[r];
      O[(size_t)rowg * DIM + h * 64 + t * 16 + ln] = (bf16_t)val;
    }
}

// ---- launcher ------------------------------------------------------------
extern "C" void kernel_launch(void* const* d_in, const int* in_sizes, int n_in,
                              void* d_out, int out_size, void* d_ws, size_t ws_size,
                              hipStream_t stream) {
  const float* query = (const float*)d_in[0];
  const float* key   = (const float*)d_in[1];
  const float* value = (const float*)d_in[2];
  const float* Wq = (const float*)d_in[3];  const float* bq = (const float*)d_in[4];
  const float* Wk = (const float*)d_in[5];  const float* bk = (const float*)d_in[6];
  const float* Wv = (const float*)d_in[7];  const float* bv = (const float*)d_in[8];
  const float* Wo = (const float*)d_in[9];  const float* bo = (const float*)d_in[10];
  float* out = (float*)d_out;

  bf16_t* Qb = (bf16_t*)d_ws;                       // NSEQ x DIM
  bf16_t* Kb = Qb + (size_t)NSEQ * DIM;             // NSEQ x KVD
  bf16_t* Vb = Kb + (size_t)NSEQ * KVD;             // NSEQ x KVD
  bf16_t* Ab = Vb + (size_t)NSEQ * KVD;             // NSEQ x DIM (attention out)

  // projections (bf16 out)
  gemm_bias_kernel<float, bf16_t><<<dim3(DIM / 128, NSEQ / 64), 256, 0, stream>>>(
      query, Wq, bq, Qb, NSEQ, DIM, DIM);
  gemm_bias_kernel<float, bf16_t><<<dim3(KVD / 128, NSEQ / 64), 256, 0, stream>>>(
      key, Wk, bk, Kb, NSEQ, KVD, DIM);
  gemm_bias_kernel<float, bf16_t><<<dim3(KVD / 128, NSEQ / 64), 256, 0, stream>>>(
      value, Wv, bv, Vb, NSEQ, KVD, DIM);

  // RoPE on Q and K
  {
    int nq = NSEQ * (DIM / 2);
    int nk = NSEQ * (KVD / 2);
    rope_kernel<<<(nq + 255) / 256, 256, 0, stream>>>(Qb, DIM);
    rope_kernel<<<(nk + 255) / 256, 256, 0, stream>>>(Kb, KVD);
  }

  // flash attention
  attn_kernel<<<dim3(NSEQ / 64, QH), 128, 0, stream>>>(Qb, Kb, Vb, Ab);

  // output projection (f32 out)
  gemm_bias_kernel<bf16_t, float><<<dim3(DIM / 128, NSEQ / 64), 256, 0, stream>>>(
      Ab, Wo, bo, out, NSEQ, DIM, DIM);
}
